// GraphEncoder_6305011990764
// MI455X (gfx1250) — matrix operand order repaired and checked by reading, compiled, not run on hardware
//
#include <hip/hip_runtime.h>

// ---------- constants ----------
#define Nn   8000
#define Ee   24000
#define Bb   256
#define ATOM 133
#define BOND 14
#define Hh   256
#define HEADS 8
#define HH   2048          // HEADS * H
#define Ll   5
#define EA   (Ee + Nn)     // edges + self loops
#define NEGs 0.2f
#define EPSs 1e-5f

typedef __attribute__((ext_vector_type(16))) __bf16 v16bf;
typedef __attribute__((ext_vector_type(8)))  __bf16 bf16x8;
typedef __attribute__((ext_vector_type(8)))  float  v8f;

// ---------- helpers ----------
__device__ inline void atomicMaxF(float* addr, float val) {
    int* ai = (int*)addr;
    int old = __float_as_int(*addr);
    while (__int_as_float(old) < val) {
        int assumed = old;
        old = atomicCAS(ai, assumed, __float_as_int(val));
        if (old == assumed) break;
    }
}

__global__ __launch_bounds__(256) void k_fill(float* p, float v, long n) {
    long i = (long)blockIdx.x * 256 + threadIdx.x;
    if (i < n) p[i] = v;
}

// ---------- self-loop attr: deg + sum of incoming edge_attr ----------
__global__ __launch_bounds__(256) void k_deg(const int* __restrict__ ei,
                                             const float* __restrict__ eattr,
                                             float* deg, float* loop) {
    int e = blockIdx.x * 256 + threadIdx.x;
    if (e >= Ee) return;
    int dst = ei[Ee + e];
    atomicAdd(&deg[dst], 1.0f);
#pragma unroll
    for (int b = 0; b < BOND; ++b)
        atomicAdd(&loop[(long)dst * BOND + b], eattr[(long)e * BOND + b]);
}

__global__ __launch_bounds__(256) void k_loopdiv(float* loop, const float* deg) {
    int i = blockIdx.x * 256 + threadIdx.x;
    if (i >= Nn * BOND) return;
    loop[i] /= fmaxf(deg[i / BOND], 1.0f);
}

// ---------- atom embedding: Linear -> LayerNorm -> ReLU (block per node) ----------
__global__ __launch_bounds__(256) void k_embed(const float* __restrict__ x,
                                               const float* __restrict__ W,
                                               const float* __restrict__ b,
                                               const float* __restrict__ g,
                                               const float* __restrict__ beta,
                                               float* __restrict__ h,
                                               __bf16* __restrict__ hbf) {
    __shared__ float xs[ATOM];
    __shared__ float red[256];
    __shared__ float stat[2];
    int n = blockIdx.x, t = threadIdx.x;
    if (t < ATOM) xs[t] = x[(long)n * ATOM + t];
    __syncthreads();
    float acc = b[t];
    for (int k = 0; k < ATOM; ++k) acc += xs[k] * W[k * Hh + t];
    red[t] = acc; __syncthreads();
    for (int s = 128; s > 0; s >>= 1) { if (t < s) red[t] += red[t + s]; __syncthreads(); }
    if (t == 0) stat[0] = red[0] * (1.0f / Hh);
    __syncthreads();
    float d = acc - stat[0];
    red[t] = d * d; __syncthreads();
    for (int s = 128; s > 0; s >>= 1) { if (t < s) red[t] += red[t + s]; __syncthreads(); }
    if (t == 0) stat[1] = rsqrtf(red[0] * (1.0f / Hh) + EPSs);
    __syncthreads();
    float v = fmaxf(d * stat[1] * g[t] + beta[t], 0.0f);
    h[(long)n * Hh + t]   = v;
    hbf[(long)n * Hh + t] = (__bf16)v;
}

// ---------- pack weight [256,2048] f32 -> bf16 in WMMA B lane order ----------
// P[((kk*128 + tc)*32 + lane)*16 + j] = W[(kk*32 + klo + j)*2048 + tc*16 + l15]
__global__ __launch_bounds__(256) void k_pack_b(const float* __restrict__ W,
                                                __bf16* __restrict__ P) {
    int idx = blockIdx.x * 256 + threadIdx.x;   // 8*128*32 = 32768 threads
    int lane = idx & 31;
    int tc   = (idx >> 5) & 127;
    int kk   = idx >> 12;
    int l15  = lane & 15;
    int klo  = (lane >= 16) ? 16 : 0;
    const float* src = W + (long)(kk * 32 + klo) * HH + tc * 16 + l15;
    __bf16* dst = P + (long)idx * 16;
#pragma unroll
    for (int j = 0; j < 16; ++j) dst[j] = (__bf16)src[(long)j * HH];
}

// ---------- bf16 WMMA GEMM: C[8000,2048] = A_bf[8000,256] * B_pk ----------
// wave strip = 16 rows x 64 cols (4 col tiles); K fully unrolled -> 32 wmma
__global__ __launch_bounds__(256) void k_gemm_wmma(const __bf16* __restrict__ A,
                                                   const __bf16* __restrict__ Bp,
                                                   float* __restrict__ C) {
    int wave = threadIdx.x >> 5;
    int lane = threadIdx.x & 31;
    int task = blockIdx.x * 8 + wave;           // 500*32 = 16000 tasks, grid=2000
    int tr = task >> 5;                          // row tile 0..499
    int cs = task & 31;                          // col strip 0..31 (64 cols)
    bool hi = lane >= 16;
    int  l15 = lane & 15;
    int  off1 = hi ? 16 : 0, off2 = hi ? 24 : 8; // A K-offsets per lane half
    const __bf16* arow = A + (long)(tr * 16 + l15) * Hh;
    v8f a0 = {}, a1 = {}, a2 = {}, a3 = {};
#pragma unroll
    for (int kk = 0; kk < 8; ++kk) {
        bf16x8 alo = *(const bf16x8*)(arow + kk * 32 + off1);
        bf16x8 ahi = *(const bf16x8*)(arow + kk * 32 + off2);
        v16bf a;
#pragma unroll
        for (int j = 0; j < 8; ++j) { a[j] = alo[j]; a[8 + j] = ahi[j]; }
        const __bf16* bp = Bp + ((long)(kk * 128 + cs * 4) * 32 + lane) * 16;
        v16bf b0 = *(const v16bf*)(bp);
        v16bf b1 = *(const v16bf*)(bp + 512);
        v16bf b2 = *(const v16bf*)(bp + 1024);
        v16bf b3 = *(const v16bf*)(bp + 1536);
        a0 = __builtin_amdgcn_wmma_f32_16x16x32_bf16(false, a, false, b0, (short)0, a0, false, false);
        a1 = __builtin_amdgcn_wmma_f32_16x16x32_bf16(false, a, false, b1, (short)0, a1, false, false);
        a2 = __builtin_amdgcn_wmma_f32_16x16x32_bf16(false, a, false, b2, (short)0, a2, false, false);
        a3 = __builtin_amdgcn_wmma_f32_16x16x32_bf16(false, a, false, b3, (short)0, a3, false, false);
    }
    int rbase = tr * 16 + (hi ? 8 : 0);
    int cbase = cs * 64 + l15;
#pragma unroll
    for (int i = 0; i < 8; ++i) {
        float* crow = C + (long)(rbase + i) * HH + cbase;
        crow[0]  = a0[i];
        crow[16] = a1[i];
        crow[32] = a2[i];
        crow[48] = a3[i];
    }
}

// ---------- attention scores: block per edge, wave per head ----------
__global__ __launch_bounds__(256) void k_score(const float* __restrict__ xl,
                                               const float* __restrict__ xr,
                                               const float* __restrict__ eattr,
                                               const float* __restrict__ loop,
                                               const float* __restrict__ We_l,
                                               const float* __restrict__ att_l,
                                               const int* __restrict__ ei,
                                               float* __restrict__ score) {
    __shared__ float eas[BOND];
    int e = blockIdx.x, t = threadIdx.x;
    int src, dst;
    const float* ea;
    if (e < Ee) { src = ei[e]; dst = ei[Ee + e]; ea = eattr + (long)e * BOND; }
    else        { src = dst = e - Ee;            ea = loop  + (long)(e - Ee) * BOND; }
    if (t < BOND) eas[t] = ea[t];
    __syncthreads();
    int hd = t >> 5, lane = t & 31;
    const float* xls  = xl + (long)src * HH + hd * Hh;
    const float* xrs  = xr + (long)dst * HH + hd * Hh;
    const float* attp = att_l + hd * Hh;
    const float* wep  = We_l + hd * Hh;
    float acc = 0.0f;
#pragma unroll
    for (int j = 0; j < 8; ++j) {
        int d = lane + j * 32;
        float ee = 0.0f;
#pragma unroll
        for (int b = 0; b < BOND; ++b) ee += eas[b] * wep[b * HH + d];
        float f = xls[d] + xrs[d] + ee;
        f = f > 0.0f ? f : NEGs * f;
        acc += f * attp[d];
    }
#pragma unroll
    for (int m = 16; m > 0; m >>= 1) acc += __shfl_xor(acc, m, 32);
    if (lane == 0) score[(long)e * HEADS + hd] = acc;
}

// ---------- scatter softmax over destination ----------
__global__ __launch_bounds__(256) void k_segmax(const float* __restrict__ score,
                                                const int* __restrict__ ei, float* mbuf) {
    int idx = blockIdx.x * 256 + threadIdx.x;
    if (idx >= EA * HEADS) return;
    int e = idx >> 3, hd = idx & 7;
    int dst = (e < Ee) ? ei[Ee + e] : (e - Ee);
    atomicMaxF(&mbuf[dst * HEADS + hd], score[idx]);
}

__global__ __launch_bounds__(256) void k_expsum(float* __restrict__ score,
                                                const int* __restrict__ ei,
                                                const float* __restrict__ mbuf,
                                                float* __restrict__ sbuf) {
    int idx = blockIdx.x * 256 + threadIdx.x;
    if (idx >= EA * HEADS) return;
    int e = idx >> 3, hd = idx & 7;
    int dst = (e < Ee) ? ei[Ee + e] : (e - Ee);
    float ex = __expf(score[idx] - mbuf[dst * HEADS + hd]);
    score[idx] = ex;
    atomicAdd(&sbuf[dst * HEADS + hd], ex);
}

// ---------- message accumulation + head mean (block per edge) ----------
__global__ __launch_bounds__(256) void k_msg(const float* __restrict__ xl,
                                             const float* __restrict__ ex,
                                             const float* __restrict__ sbuf,
                                             const int* __restrict__ ei,
                                             float* __restrict__ outb) {
    __shared__ float al[HEADS];
    int e = blockIdx.x, t = threadIdx.x;
    int src, dst;
    if (e < Ee) { src = ei[e]; dst = ei[Ee + e]; } else { src = dst = e - Ee; }
    if (t < HEADS) al[t] = ex[(long)e * HEADS + t] / sbuf[dst * HEADS + t];
    __syncthreads();
    float acc = 0.0f;
#pragma unroll
    for (int hd = 0; hd < HEADS; ++hd)
        acc += al[hd] * xl[(long)src * HH + hd * Hh + t];
    atomicAdd(&outb[(long)dst * Hh + t], 0.125f * acc);
}

// ---------- residual + LayerNorm (block per node) ----------
__global__ __launch_bounds__(256) void k_lnres(const float* __restrict__ outb,
                                               const float* __restrict__ bias,
                                               const float* __restrict__ hc,
                                               const float* __restrict__ g,
                                               const float* __restrict__ b2,
                                               float* __restrict__ hn,
                                               __bf16* __restrict__ hbf) {
    __shared__ float red[256];
    __shared__ float stat[2];
    int n = blockIdx.x, t = threadIdx.x;
    float acc = outb[(long)n * Hh + t] + bias[t] + hc[(long)n * Hh + t];
    red[t] = acc; __syncthreads();
    for (int s = 128; s > 0; s >>= 1) { if (t < s) red[t] += red[t + s]; __syncthreads(); }
    if (t == 0) stat[0] = red[0] * (1.0f / Hh);
    __syncthreads();
    float d = acc - stat[0];
    red[t] = d * d; __syncthreads();
    for (int s = 128; s > 0; s >>= 1) { if (t < s) red[t] += red[t + s]; __syncthreads(); }
    if (t == 0) stat[1] = rsqrtf(red[0] * (1.0f / Hh) + EPSs);
    __syncthreads();
    float v = d * stat[1] * g[t] + b2[t];
    hn[(long)n * Hh + t]  = v;
    hbf[(long)n * Hh + t] = (__bf16)v;
}

// ---------- readout gate: relu(h@g1W+g1b)@g2W+g2b ----------
__global__ __launch_bounds__(128) void k_gate(const float* __restrict__ h,
                                              const float* __restrict__ g1W,
                                              const float* __restrict__ g1b,
                                              const float* __restrict__ g2W,
                                              const float* __restrict__ g2b,
                                              float* __restrict__ gate) {
    __shared__ float hs[Hh];
    __shared__ float red[128];
    int n = blockIdx.x, t = threadIdx.x;
    hs[t]       = h[(long)n * Hh + t];
    hs[t + 128] = h[(long)n * Hh + t + 128];
    __syncthreads();
    float acc = g1b[t];
    for (int k = 0; k < Hh; ++k) acc += hs[k] * g1W[k * 128 + t];
    acc = fmaxf(acc, 0.0f) * g2W[t];
    red[t] = acc; __syncthreads();
    for (int s = 64; s > 0; s >>= 1) { if (t < s) red[t] += red[t + s]; __syncthreads(); }
    if (t == 0) gate[n] = red[0] + g2b[0];
}

__global__ __launch_bounds__(256) void k_bmax(const float* gate, const int* bv, float* gm) {
    int n = blockIdx.x * 256 + threadIdx.x;
    if (n >= Nn) return;
    atomicMaxF(&gm[bv[n]], gate[n]);
}

__global__ __launch_bounds__(256) void k_bexp(float* gate, const int* bv,
                                              const float* gm, float* gs) {
    int n = blockIdx.x * 256 + threadIdx.x;
    if (n >= Nn) return;
    float ge = __expf(gate[n] - gm[bv[n]]);
    gate[n] = ge;
    atomicAdd(&gs[bv[n]], ge);
}

__global__ __launch_bounds__(256) void k_final(const float* __restrict__ h,
                                               const float* __restrict__ gate,
                                               const float* __restrict__ gs,
                                               const int* __restrict__ bv,
                                               float* __restrict__ out) {
    int n = blockIdx.x, t = threadIdx.x;
    int b = bv[n];
    float w = gate[n] / fmaxf(gs[b], 1e-16f);
    atomicAdd(&out[(long)b * Hh + t], w * h[(long)n * Hh + t]);
}

// ---------- host ----------
extern "C" void kernel_launch(void* const* d_in, const int* in_sizes, int n_in,
                              void* d_out, int out_size, void* d_ws, size_t ws_size,
                              hipStream_t stream) {
    const float* x      = (const float*)d_in[0];
    const float* eattr  = (const float*)d_in[1];
    const float* emb_W  = (const float*)d_in[2];
    const float* emb_b  = (const float*)d_in[3];
    const float* emb_g  = (const float*)d_in[4];
    const float* emb_be = (const float*)d_in[5];
    const float* Wl     = (const float*)d_in[6];
    const float* Wr     = (const float*)d_in[7];
    const float* We     = (const float*)d_in[8];
    const float* att    = (const float*)d_in[9];
    const float* bias   = (const float*)d_in[10];
    const float* ln_g   = (const float*)d_in[11];
    const float* ln_b   = (const float*)d_in[12];
    const float* g1W    = (const float*)d_in[13];
    const float* g1b    = (const float*)d_in[14];
    const float* g2W    = (const float*)d_in[15];
    const float* g2b    = (const float*)d_in[16];
    const int*   ei     = (const int*)d_in[17];
    const int*   bv     = (const int*)d_in[18];
    float* out = (float*)d_out;

    float* ws   = (float*)d_ws;
    float* deg  = ws;  ws += Nn;
    float* loop = ws;  ws += (long)Nn * BOND;
    float* h0   = ws;  ws += (long)Nn * Hh;
    float* h1   = ws;  ws += (long)Nn * Hh;
    float* xl   = ws;  ws += (long)Nn * HH;
    float* xr   = ws;  ws += (long)Nn * HH;
    float* sc   = ws;  ws += (long)EA * HEADS;
    float* mbuf = ws;  ws += (long)Nn * HEADS;
    float* sbuf = ws;  ws += (long)Nn * HEADS;
    float* outb = ws;  ws += (long)Nn * Hh;
    float* gate = ws;  ws += Nn;
    float* gm   = ws;  ws += Bb;
    float* gs   = ws;  ws += Bb;
    __bf16* hbf0 = (__bf16*)ws;  ws += (long)Nn * Hh / 2;   // bf16 copy of h (ping)
    __bf16* hbf1 = (__bf16*)ws;  ws += (long)Nn * Hh / 2;   // bf16 copy of h (pong)
    __bf16* bpk  = (__bf16*)ws;  ws += (long)8 * 128 * 32 * 16 / 2;  // packed weight

    auto blocks = [](long n) { return (int)((n + 255) / 256); };

    // self-loop attributes
    k_fill<<<blocks(Nn), 256, 0, stream>>>(deg, 0.0f, Nn);
    k_fill<<<blocks((long)Nn * BOND), 256, 0, stream>>>(loop, 0.0f, (long)Nn * BOND);
    k_deg<<<blocks(Ee), 256, 0, stream>>>(ei, eattr, deg, loop);
    k_loopdiv<<<blocks((long)Nn * BOND), 256, 0, stream>>>(loop, deg);

    // embedding
    k_embed<<<Nn, 256, 0, stream>>>(x, emb_W, emb_b, emb_g, emb_be, h0, hbf0);

    float*  hc  = h0;   float*  hn  = h1;
    __bf16* hbc = hbf0; __bf16* hbn = hbf1;
    const int gemmBlocks = (500 * 32) / 8;    // 16000 wave-strips, 8 waves/block
    const int packBlocks = (8 * 128 * 32) / 256;

    for (int l = 0; l < Ll; ++l) {
        const float* Wl_l = Wl + (long)l * Hh * HH;
        const float* Wr_l = Wr + (long)l * Hh * HH;
        const float* We_l = We + (long)l * BOND * HH;
        const float* at_l = att + (long)l * HEADS * Hh;
        const float* bi_l = bias + (long)l * Hh;
        const float* lg_l = ln_g + (long)l * Hh;
        const float* lb_l = ln_b + (long)l * Hh;

        k_pack_b<<<packBlocks, 256, 0, stream>>>(Wl_l, bpk);
        k_gemm_wmma<<<gemmBlocks, 256, 0, stream>>>(hbc, bpk, xl);
        k_pack_b<<<packBlocks, 256, 0, stream>>>(Wr_l, bpk);
        k_gemm_wmma<<<gemmBlocks, 256, 0, stream>>>(hbc, bpk, xr);

        k_score<<<EA, 256, 0, stream>>>(xl, xr, eattr, loop, We_l, at_l, ei, sc);

        k_fill<<<blocks((long)Nn * HEADS), 256, 0, stream>>>(mbuf, -3.0e38f, (long)Nn * HEADS);
        k_fill<<<blocks((long)Nn * HEADS), 256, 0, stream>>>(sbuf, 0.0f, (long)Nn * HEADS);
        k_segmax<<<blocks((long)EA * HEADS), 256, 0, stream>>>(sc, ei, mbuf);
        k_expsum<<<blocks((long)EA * HEADS), 256, 0, stream>>>(sc, ei, mbuf, sbuf);

        k_fill<<<blocks((long)Nn * Hh), 256, 0, stream>>>(outb, 0.0f, (long)Nn * Hh);
        k_msg<<<EA, 256, 0, stream>>>(xl, sc, sbuf, ei, outb);

        k_lnres<<<Nn, 256, 0, stream>>>(outb, bi_l, hc, lg_l, lb_l, hn, hbn);
        float* tf = hc; hc = hn; hn = tf;
        __bf16* tb = hbc; hbc = hbn; hbn = tb;
    }

    // GlobalAttention readout
    k_gate<<<Nn, 128, 0, stream>>>(hc, g1W, g1b, g2W, g2b, gate);
    k_fill<<<blocks(Bb), 256, 0, stream>>>(gm, -3.0e38f, Bb);
    k_fill<<<blocks(Bb), 256, 0, stream>>>(gs, 0.0f, Bb);
    k_fill<<<blocks((long)Bb * Hh), 256, 0, stream>>>(out, 0.0f, (long)Bb * Hh);
    k_bmax<<<blocks(Nn), 256, 0, stream>>>(gate, bv, gm);
    k_bexp<<<blocks(Nn), 256, 0, stream>>>(gate, bv, gm, gs);
    k_final<<<Nn, 256, 0, stream>>>(hc, gate, gs, bv, out);

    (void)in_sizes; (void)n_in; (void)out_size; (void)ws_size;
}
